// ClassicMambaBlock_74741020885095
// MI455X (gfx1250) — matrix operand order repaired and checked by reading, not compiled, not run on hardware
//
#include <hip/hip_runtime.h>
#include <hip/hip_bf16.h>
#include <math.h>

typedef _Float16 v8h  __attribute__((ext_vector_type(8)));
typedef _Float16 v16h __attribute__((ext_vector_type(16)));
typedef float    v8f  __attribute__((ext_vector_type(8)));

#define DEVI __device__ __forceinline__

enum { EPI_STORE = 0, EPI_SOFTPLUS_BIAS = 1 };

// ---------------------------------------------------------------------------
// C[M,N] = A[M,K] @ W[N,K]^T   (A row-major stride lda, W row-major stride ldb)
//
// Block tile 128x256, 256 threads = 8 waves arranged 2 (M) x 4 (N);
// each wave owns a 64x64 subtile = 4x4 wmma accumulators (128 VGPRs).
// Out-of-range column tiles read from a zero-filled scratch page so the
// k-loop is branchless: global_load_b128 + v_wmma only.
//
// Fragment layouts used here (wave32, contiguous k, same order for A and B):
//  A 16x32 f16: lane row M=lane%16; K = k0 + (lane/16)*16 + i  -> one 32B load.
//  B 32x16 f16: lane col N=lane%16; K = k0 + (lane/16)*16 + i  -> one 32B load.
// ---------------------------------------------------------------------------
template <typename OutT, int EPI>
__global__ __launch_bounds__(256)
void wmma_gemm_kernel(const _Float16* __restrict__ A, int lda,
                      const _Float16* __restrict__ W, int ldb,
                      OutT* __restrict__ C, int ldc,
                      int M, int N, int K,
                      const float* __restrict__ bias,
                      const _Float16* __restrict__ zbuf) {
  const int lane = threadIdx.x & 31;
  const int wave = threadIdx.x >> 5;
  const int half = lane >> 4;   // which 16-lane half
  const int l15  = lane & 15;
  const int wm = wave & 1, wn = wave >> 1;
  const int row0 = blockIdx.y * 128 + wm * 64;
  const int col0 = blockIdx.x * 256 + wn * 64;

  // loop-invariant fragment base pointers (OOB columns -> zero page)
  const _Float16* abase[4];
  const _Float16* bbase[4];
#pragma unroll
  for (int i = 0; i < 4; ++i)
    abase[i] = A + (size_t)(row0 + i * 16 + l15) * lda + half * 16;
#pragma unroll
  for (int j = 0; j < 4; ++j) {
    int n = col0 + j * 16 + l15;
    bbase[j] = (n < N) ? (W + (size_t)n * ldb + half * 16) : (zbuf + half * 16);
  }

  v8f acc[4][4] = {};

  for (int k0 = 0; k0 < K; k0 += 32) {
    if (k0 + 32 < K) {  // gfx1250 global_prefetch_b8 hint for next K slice
      __builtin_prefetch(abase[0] + k0 + 32, 0, 3);
    }
    v16h a[4], b[4];
#pragma unroll
    for (int i = 0; i < 4; ++i) {
      a[i] = *(const v16h*)(abase[i] + k0);
    }
#pragma unroll
    for (int j = 0; j < 4; ++j)
      b[j] = *(const v16h*)(bbase[j] + k0);
#pragma unroll
    for (int i = 0; i < 4; ++i)
#pragma unroll
      for (int j = 0; j < 4; ++j)
        acc[i][j] = __builtin_amdgcn_wmma_f32_16x16x32_f16(
            false, a[i], false, b[j], (short)0, acc[i][j], false, false);
  }

  // C/D layout: VGPR v, lanes 0-15 -> M = v ; lanes 16-31 -> M = 8+v ; N = lane%16
#pragma unroll
  for (int tm = 0; tm < 4; ++tm) {
#pragma unroll
    for (int tn = 0; tn < 4; ++tn) {
      int c = col0 + tn * 16 + l15;
      if (c >= N) continue;
      int rbase = row0 + tm * 16 + half * 8;
#pragma unroll
      for (int v = 0; v < 8; ++v) {
        float val = acc[tm][tn][v];
        if (EPI == EPI_SOFTPLUS_BIAS) {
          float t = val + bias[c];
          val = (t > 20.0f) ? t : log1pf(__expf(t));
        }
        C[(size_t)(rbase + v) * ldc + c] = (OutT)val;
      }
    }
  }
}

// ---------------------------------------------------------------------------
__global__ void cvt_f32_f16_kernel(const float* __restrict__ in,
                                   _Float16* __restrict__ out, size_t n) {
  size_t i = (size_t)blockIdx.x * blockDim.x + threadIdx.x;
  if (i < n) out[i] = (_Float16)in[i];
}

__global__ void zero_f16_kernel(_Float16* __restrict__ p, size_t n) {
  size_t i = (size_t)blockIdx.x * blockDim.x + threadIdx.x;
  if (i < n) p[i] = (_Float16)0.0f;
}

// causal depthwise conv1d (K=3) + SiLU; reads xc = first E columns of xz.
__global__ void conv_silu_kernel(const _Float16* __restrict__ xz,
                                 const float* __restrict__ conv_w,
                                 const float* __restrict__ conv_b,
                                 _Float16* __restrict__ u,
                                 int L, int E, size_t total) {
  size_t i = (size_t)blockIdx.x * blockDim.x + threadIdx.x;
  if (i >= total) return;
  int e = (int)(i % E);
  size_t row = i / E;          // b*L + l
  int l = (int)(row % L);
  const int twoE = 2 * E;
  float acc = conv_b[e];
#pragma unroll
  for (int k = 0; k < 3; ++k) {
    int t = l + k - 2;
    if (t >= 0) acc += conv_w[e * 3 + k] * (float)xz[(row + (size_t)(k - 2)) * twoE + e];
  }
  float s = acc / (1.0f + __expf(-acc));   // SiLU
  u[i] = (_Float16)s;
}

// selective scan: one lane per (b,e,n) state; 16-lane subgroup reduction.
__global__ void scan_kernel(const float* __restrict__ delta,
                            const _Float16* __restrict__ u,
                            const _Float16* __restrict__ dbc,
                            const float* __restrict__ A_log,
                            float* __restrict__ ys,
                            int L, int E, int BE) {
  int t = blockIdx.x * blockDim.x + threadIdx.x;
  int n = t & 15;
  int g = t >> 4;              // (b,e) group
  if (g >= BE) return;
  int e = g % E;
  int b = g / E;
  float A = -__expf(A_log[e * 16 + n]);
  float h = 0.0f;
  size_t base = (size_t)b * L;
  for (int step = 0; step < L; ++step) {
    size_t row = base + step;
    float d  = delta[row * E + e];
    float uu = (float)u[row * E + e];
    float Bv = (float)dbc[row * 96 + 64 + n];
    float Cv = (float)dbc[row * 96 + 80 + n];
    h = __expf(d * A) * h + d * uu * Bv;
    float p = h * Cv;
#pragma unroll
    for (int m = 8; m >= 1; m >>= 1) p += __shfl_xor(p, m, 16);
    if (n == 0) ys[row * E + e] = p;
  }
}

// y = (ys + u*D) * silu(z), output f16 for the final WMMA GEMM.
__global__ void gate_kernel(const float* __restrict__ ys,
                            const _Float16* __restrict__ u,
                            const _Float16* __restrict__ xz,
                            const float* __restrict__ D_par,
                            _Float16* __restrict__ yg,
                            int E, size_t total) {
  size_t i = (size_t)blockIdx.x * blockDim.x + threadIdx.x;
  if (i >= total) return;
  int e = (int)(i % E);
  size_t row = i / E;
  float z = (float)xz[row * (size_t)(2 * E) + E + e];
  float v = (ys[i] + (float)u[i] * D_par[e]) * (z / (1.0f + __expf(-z)));
  yg[i] = (_Float16)v;
}

// ---------------------------------------------------------------------------
extern "C" void kernel_launch(void* const* d_in, const int* in_sizes, int n_in,
                              void* d_out, int out_size, void* d_ws, size_t ws_size,
                              hipStream_t stream) {
  (void)in_sizes; (void)n_in; (void)out_size; (void)ws_size;
  const float* x      = (const float*)d_in[0];
  const float* W_in   = (const float*)d_in[1];
  const float* conv_w = (const float*)d_in[2];
  const float* conv_b = (const float*)d_in[3];
  const float* W_sel  = (const float*)d_in[4];
  const float* dt_w   = (const float*)d_in[5];
  const float* dt_b   = (const float*)d_in[6];
  const float* A_log  = (const float*)d_in[7];
  const float* D_par  = (const float*)d_in[8];
  const float* W_out  = (const float*)d_in[9];
  float* out = (float*)d_out;

  constexpr int B = 2, L = 2048, Dm = 1024, E = 2048, NST = 16, R = 64;
  constexpr int M = B * L;                 // 4096 token rows
  constexpr int twoE = 2 * E;              // 4096
  constexpr int NSEL = R + 2 * NST;        // 96
  constexpr int ZN = 4096;                 // zero page (covers any K slice)

  // workspace arena (f16/f32 intermediates, ~157 MB)
  char* p = (char*)d_ws;
  auto arena = [&](size_t bytes) {
    char* q = p;
    p += (bytes + 255) & ~(size_t)255;
    return q;
  };
  _Float16* zbuf   = (_Float16*)arena((size_t)ZN * 2);
  _Float16* x_h    = (_Float16*)arena((size_t)M * Dm * 2);
  _Float16* Win_h  = (_Float16*)arena((size_t)twoE * Dm * 2);
  _Float16* Wsel_h = (_Float16*)arena((size_t)NSEL * E * 2);
  _Float16* dtw_h  = (_Float16*)arena((size_t)E * R * 2);
  _Float16* Wout_h = (_Float16*)arena((size_t)Dm * E * 2);
  _Float16* xz_h   = (_Float16*)arena((size_t)M * twoE * 2);
  _Float16* u_h    = (_Float16*)arena((size_t)M * E * 2);
  _Float16* dbc_h  = (_Float16*)arena((size_t)M * NSEL * 2);
  float*    delta  = (float*)   arena((size_t)M * E * 4);
  float*    ys     = (float*)   arena((size_t)M * E * 4);
  _Float16* yg_h   = (_Float16*)arena((size_t)M * E * 2);

  auto blocks = [](size_t n) { return (unsigned)((n + 255) / 256); };

  // 0) zero page for branchless OOB column tiles
  zero_f16_kernel<<<blocks(ZN), 256, 0, stream>>>(zbuf, (size_t)ZN);

  // 1) f32 -> f16 conversions of GEMM operands
  cvt_f32_f16_kernel<<<blocks((size_t)M * Dm),    256, 0, stream>>>(x,     x_h,    (size_t)M * Dm);
  cvt_f32_f16_kernel<<<blocks((size_t)twoE * Dm), 256, 0, stream>>>(W_in,  Win_h,  (size_t)twoE * Dm);
  cvt_f32_f16_kernel<<<blocks((size_t)NSEL * E),  256, 0, stream>>>(W_sel, Wsel_h, (size_t)NSEL * E);
  cvt_f32_f16_kernel<<<blocks((size_t)E * R),     256, 0, stream>>>(dt_w,  dtw_h,  (size_t)E * R);
  cvt_f32_f16_kernel<<<blocks((size_t)Dm * E),    256, 0, stream>>>(W_out, Wout_h, (size_t)Dm * E);

  // 2) in-projection: xz[M,2E] = x @ W_in^T   (34.4 GFLOP, WMMA)
  wmma_gemm_kernel<_Float16, EPI_STORE>
      <<<dim3(twoE / 256, M / 128), 256, 0, stream>>>(
          x_h, Dm, Win_h, Dm, xz_h, twoE, M, twoE, Dm, nullptr, zbuf);

  // 3) causal depthwise conv + SiLU -> u
  conv_silu_kernel<<<blocks((size_t)M * E), 256, 0, stream>>>(
      xz_h, conv_w, conv_b, u_h, L, E, (size_t)M * E);

  // 4) selective projection: dbc[M,96] = u @ W_sel^T  (WMMA, ragged N)
  wmma_gemm_kernel<_Float16, EPI_STORE>
      <<<dim3(1, M / 128), 256, 0, stream>>>(
          u_h, E, Wsel_h, E, dbc_h, NSEL, M, NSEL, E, nullptr, zbuf);

  // 5) delta[M,E] = softplus(dt_low @ dt_w^T + dt_b)  (WMMA, fused epilogue)
  //    dt_low = first 64 columns of dbc (lda = 96)
  wmma_gemm_kernel<float, EPI_SOFTPLUS_BIAS>
      <<<dim3(E / 256, M / 128), 256, 0, stream>>>(
          dbc_h, NSEL, dtw_h, R, delta, E, M, E, R, dt_b, zbuf);

  // 6) selective scan (sequential in L, parallel over B*E*N states)
  scan_kernel<<<(B * E * NST) / 256, 256, 0, stream>>>(
      delta, u_h, dbc_h, A_log, ys, L, E, B * E);

  // 7) gating: yg = (ys + u*D) * silu(z)
  gate_kernel<<<blocks((size_t)M * E), 256, 0, stream>>>(
      ys, u_h, xz_h, D_par, yg_h, E, (size_t)M * E);

  // 8) out-projection: out[M,Dm] = yg @ W_out^T  (17.2 GFLOP, WMMA)
  wmma_gemm_kernel<float, EPI_STORE>
      <<<dim3(Dm / 256, M / 128), 256, 0, stream>>>(
          yg_h, E, Wout_h, E, out, Dm, M, Dm, E, nullptr, zbuf);
}
